// vposer_decoder_57329223467193
// MI455X (gfx1250) — matrix operand order repaired and checked by reading, not compiled, not run on hardware
//
#include <hip/hip_runtime.h>
#include <hip/hip_bf16.h>
#include <math.h>

typedef _Float16 v16h __attribute__((ext_vector_type(16)));
typedef _Float16 v8h  __attribute__((ext_vector_type(8)));
typedef float    v8f  __attribute__((ext_vector_type(8)));

#define B_ROWS 32768
#define D_IN   32
#define H_DIM  512
#define N_OUT  207
#define NP3    256   // N padded to tile multiple for GEMM3

// ---------------------------------------------------------------------------
// f32 -> f16 conversion (for x)
// ---------------------------------------------------------------------------
__global__ __launch_bounds__(256)
void f32_to_f16_kernel(const float* __restrict__ src, _Float16* __restrict__ dst, int n) {
  int t = blockIdx.x * blockDim.x + threadIdx.x;
  if (t < n) dst[t] = (_Float16)src[t];
}

// ---------------------------------------------------------------------------
// Pack weight matrix W (K x N, row-major f32) into WMMA B-fragment layout:
// per (ktile=32, ntile=16) fragment: 32 lanes x 16 f16, contiguous (1KB/frag).
// Lane L holds column n = ntile*16 + (L&15), K elements (e>>3)*16 + (L>>4)*8 + (e&7).
// ---------------------------------------------------------------------------
__global__ __launch_bounds__(256)
void pack_w_kernel(const float* __restrict__ W, _Float16* __restrict__ Bp,
                   int K, int N, int Np) {
  int t = blockIdx.x * blockDim.x + threadIdx.x;
  int total = (K >> 5) * (Np >> 4) * 512;
  if (t >= total) return;
  int e    = t & 15;
  int l    = (t >> 4) & 31;
  int frag = t >> 9;
  int ntiles = Np >> 4;
  int ktile = frag / ntiles;
  int ntile = frag - ktile * ntiles;
  int n = ntile * 16 + (l & 15);
  int k = ktile * 32 + (e >> 3) * 16 + (l >> 4) * 8 + (e & 7);
  float v = (k < K && n < N) ? W[(size_t)k * N + n] : 0.0f;
  Bp[t] = (_Float16)v;
}

// ---------------------------------------------------------------------------
// WMMA GEMM: C(MxN) = A(MxK,f16,row-major) * Bp(packed f16) + bias, epilogue.
// Block = 128 threads (4 waves as 2x2). Block tile 64(M) x 128(N).
// Wave tile 32(M) x 64(N) = 2x4 fragments of v_wmma_f32_16x16x32_f16.
// EPI==0: leaky-relu(0.2), store f16.  EPI==1: tanh, store f32, mask n<Nreal.
// ---------------------------------------------------------------------------
template<int EPI>
__global__ __launch_bounds__(128)
void wmma_gemm_kernel(const _Float16* __restrict__ A,
                      const _Float16* __restrict__ Bp,
                      const float* __restrict__ bias,
                      void* __restrict__ Out,
                      int K, int Np, int Nreal, int ldo) {
  const int lane = threadIdx.x & 31;
  const int wid  = threadIdx.x >> 5;
  const int wm   = wid >> 1;
  const int wn   = wid & 1;
  const int m0 = blockIdx.y * 64 + wm * 32;
  const int n0 = blockIdx.x * 128 + wn * 64;
  const int lm = lane & 15;   // A: row within tile / C,D: column / B: column
  const int lh = lane >> 4;   // half-wave select
  const int ntilesTotal = Np >> 4;
  const int ntile0 = n0 >> 4;

  v8f acc[2][4] = {};

  for (int k0 = 0; k0 < K; k0 += 32) {
    // A fragments: lane holds row m0+mi*16+lm; elems 0..7 = K lh*8.., 8..15 = K 16+lh*8..
    v16h afrag[2];
#pragma unroll
    for (int mi = 0; mi < 2; ++mi) {
      const _Float16* ap = A + (size_t)(m0 + mi * 16 + lm) * K + k0 + lh * 8;
      v8h lo = *(const v8h*)ap;
      v8h hi = *(const v8h*)(ap + 16);
      afrag[mi] = __builtin_shufflevector(lo, hi,
                    0,1,2,3,4,5,6,7,8,9,10,11,12,13,14,15);
    }
    const int ktile = k0 >> 5;
#pragma unroll
    for (int ni = 0; ni < 4; ++ni) {
      const _Float16* bp = Bp
          + (((size_t)(ktile * ntilesTotal + ntile0 + ni)) << 9) + lane * 16;
      v16h bfrag = *(const v16h*)bp;
#pragma unroll
      for (int mi = 0; mi < 2; ++mi) {
        acc[mi][ni] = __builtin_amdgcn_wmma_f32_16x16x32_f16(
            false, afrag[mi], false, bfrag, (short)0, acc[mi][ni], false, false);
      }
    }
  }

  // Epilogue. C/D layout: lane -> n = lm, vgpr r -> m = r + lh*8.
#pragma unroll
  for (int mi = 0; mi < 2; ++mi) {
#pragma unroll
    for (int ni = 0; ni < 4; ++ni) {
      const int nn = n0 + ni * 16 + lm;
      const float bv = (nn < Nreal) ? bias[nn] : 0.0f;
#pragma unroll
      for (int r = 0; r < 8; ++r) {
        const int mm = m0 + mi * 16 + lh * 8 + r;
        float v = acc[mi][ni][r] + bv;
        if (EPI == 0) {
          v = (v > 0.0f) ? v : 0.2f * v;               // leaky_relu 0.2
          ((_Float16*)Out)[(size_t)mm * ldo + nn] = (_Float16)v;
        } else {
          if (nn < Nreal)
            ((float*)Out)[(size_t)mm * ldo + nn] = tanhf(v);
        }
      }
    }
  }
}

// ---------------------------------------------------------------------------
// Per 3x3 matrix: Newton polar decomposition (-> U@V^T), then torchgeometry
// rotation_matrix_to_quaternion (on transposed matrix) + quaternion_to_angle_axis.
// ---------------------------------------------------------------------------
__global__ __launch_bounds__(256)
void polar_aa_kernel(const float* __restrict__ O, float* __restrict__ out, int total) {
  int t = blockIdx.x * blockDim.x + threadIdx.x;
  if (t >= total) return;
  int row = t / 23;
  int g = t - row * 23;
  const float* p = O + (size_t)row * 207 + g * 9;
  float x00 = p[0], x01 = p[1], x02 = p[2];
  float x10 = p[3], x11 = p[4], x12 = p[5];
  float x20 = p[6], x21 = p[7], x22 = p[8];

  // Newton iteration: X <- 0.5*(X + X^-T); X^-T = cofactor(X)/det(X).
#pragma unroll 1
  for (int it = 0; it < 12; ++it) {
    float c00 = x11 * x22 - x12 * x21;
    float c01 = x12 * x20 - x10 * x22;
    float c02 = x10 * x21 - x11 * x20;
    float c10 = x02 * x21 - x01 * x22;
    float c11 = x00 * x22 - x02 * x20;
    float c12 = x01 * x20 - x00 * x21;
    float c20 = x01 * x12 - x02 * x11;
    float c21 = x02 * x10 - x00 * x12;
    float c22 = x00 * x11 - x01 * x10;
    float det = x00 * c00 + x01 * c01 + x02 * c02;
    det = copysignf(fmaxf(fabsf(det), 1e-20f), det);
    float id = 0.5f / det;
    x00 = 0.5f * x00 + c00 * id;  x01 = 0.5f * x01 + c01 * id;  x02 = 0.5f * x02 + c02 * id;
    x10 = 0.5f * x10 + c10 * id;  x11 = 0.5f * x11 + c11 * id;  x12 = 0.5f * x12 + c12 * id;
    x20 = 0.5f * x20 + c20 * id;  x21 = 0.5f * x21 + c21 * id;  x22 = 0.5f * x22 + c22 * id;
  }

  // torchgeometry transposes first: r(i,j) = X[j][i]
  float r00 = x00, r01 = x10, r02 = x20;
  float r10 = x01, r11 = x11, r12 = x21;
  float r20 = x02, r21 = x12, r22 = x22;

  bool m2   = r22 < 1e-6f;
  bool m01  = r00 > r11;
  bool m0n1 = r00 < -r11;
  float tt, qw, qx, qy, qz;
  if (m2 && m01) {
    tt = 1.0f + r00 - r11 - r22;
    qw = r12 - r21; qx = tt;        qy = r01 + r10; qz = r20 + r02;
  } else if (m2) {
    tt = 1.0f - r00 + r11 - r22;
    qw = r20 - r02; qx = r01 + r10; qy = tt;        qz = r12 + r21;
  } else if (m0n1) {
    tt = 1.0f - r00 - r11 + r22;
    qw = r01 - r10; qx = r20 + r02; qy = r12 + r21; qz = tt;
  } else {
    tt = 1.0f + r00 + r11 + r22;
    qw = tt;        qx = r12 - r21; qy = r20 - r02; qz = r01 - r10;
  }
  float s = 0.5f * rsqrtf(fmaxf(tt, 1e-30f));
  qw *= s; qx *= s; qy *= s; qz *= s;

  float sin_sq = qx * qx + qy * qy + qz * qz;
  float sin_t = sqrtf(sin_sq);
  float cos_t = qw;
  float two_theta = 2.0f * ((cos_t < 0.0f) ? atan2f(-sin_t, -cos_t)
                                           : atan2f(sin_t, cos_t));
  float k = (sin_sq > 0.0f) ? (two_theta / sin_t) : 2.0f;

  float* o = out + (size_t)row * 69 + g * 3;
  o[0] = qx * k;
  o[1] = qy * k;
  o[2] = qz * k;
}

// ---------------------------------------------------------------------------
extern "C" void kernel_launch(void* const* d_in, const int* in_sizes, int n_in,
                              void* d_out, int out_size, void* d_ws, size_t ws_size,
                              hipStream_t stream) {
  const float* x  = (const float*)d_in[0];
  const float* W1 = (const float*)d_in[1];
  const float* b1 = (const float*)d_in[2];
  const float* W2 = (const float*)d_in[3];
  const float* b2 = (const float*)d_in[4];
  const float* W3 = (const float*)d_in[5];
  const float* b3 = (const float*)d_in[6];
  float* out = (float*)d_out;

  char* w = (char*)d_ws;
  size_t off = 0;
  auto alloc = [&](size_t bytes) -> char* {
    char* p = w + off;
    off = (off + bytes + 255) & ~(size_t)255;
    return p;
  };
  _Float16* xh  = (_Float16*)alloc((size_t)B_ROWS * D_IN * 2);           //   2 MB
  _Float16* W1p = (_Float16*)alloc((size_t)(D_IN / 32) * (H_DIM / 16) * 1024);
  _Float16* W2p = (_Float16*)alloc((size_t)(H_DIM / 32) * (H_DIM / 16) * 1024);
  _Float16* W3p = (_Float16*)alloc((size_t)(H_DIM / 32) * (NP3 / 16) * 1024);
  _Float16* H1  = (_Float16*)alloc((size_t)B_ROWS * H_DIM * 2);          //  32 MB
  _Float16* H2  = (_Float16*)alloc((size_t)B_ROWS * H_DIM * 2);          //  32 MB
  float*    O3  = (float*)H1;  // reuse H1 (dead after GEMM2): 27MB <= 32MB

  // 1) convert x to f16
  {
    int n = B_ROWS * D_IN;
    f32_to_f16_kernel<<<(n + 255) / 256, 256, 0, stream>>>(x, xh, n);
  }
  // 2) pack weights into WMMA B-fragment layout
  {
    int n1 = (D_IN / 32) * (H_DIM / 16) * 512;
    pack_w_kernel<<<(n1 + 255) / 256, 256, 0, stream>>>(W1, W1p, D_IN, H_DIM, H_DIM);
    int n2 = (H_DIM / 32) * (H_DIM / 16) * 512;
    pack_w_kernel<<<(n2 + 255) / 256, 256, 0, stream>>>(W2, W2p, H_DIM, H_DIM, H_DIM);
    int n3 = (H_DIM / 32) * (NP3 / 16) * 512;
    pack_w_kernel<<<(n3 + 255) / 256, 256, 0, stream>>>(W3, W3p, H_DIM, N_OUT, NP3);
  }
  // 3) GEMM1: (32768x32)@(32x512)+b1, leaky-relu -> H1 (f16)
  {
    dim3 grid(H_DIM / 128, B_ROWS / 64);
    wmma_gemm_kernel<0><<<grid, 128, 0, stream>>>(xh, W1p, b1, (void*)H1,
                                                  D_IN, H_DIM, H_DIM, H_DIM);
  }
  // 4) GEMM2: H1@(512x512)+b2, leaky-relu -> H2 (f16)
  {
    dim3 grid(H_DIM / 128, B_ROWS / 64);
    wmma_gemm_kernel<0><<<grid, 128, 0, stream>>>(H1, W2p, b2, (void*)H2,
                                                  H_DIM, H_DIM, H_DIM, H_DIM);
  }
  // 5) GEMM3: H2@(512x207)+b3, tanh -> O3 (f32, aliases H1)
  {
    dim3 grid(NP3 / 128, B_ROWS / 64);
    wmma_gemm_kernel<1><<<grid, 128, 0, stream>>>(H2, W3p, b3, (void*)O3,
                                                  H_DIM, NP3, N_OUT, N_OUT);
  }
  // 6) polar orthogonalization + quaternion + angle-axis -> out (32768x69)
  {
    int total = B_ROWS * 23;
    polar_aa_kernel<<<(total + 255) / 256, 256, 0, stream>>>(O3, out, total);
  }
}
